// MultiHeadedAttention_33913061769253
// MI455X (gfx1250) — matrix operand order repaired and checked by reading, not compiled, not run on hardware
//
#include <hip/hip_runtime.h>
#include <hip/hip_bf16.h>

#define D_MODEL 256
#define N_HEADS 8
#define D_K 32

typedef __bf16 bf16_t;
typedef __attribute__((ext_vector_type(16))) __bf16 v16bf;
typedef __attribute__((ext_vector_type(8)))  __bf16 v8bf;
typedef __attribute__((ext_vector_type(8)))  float  v8f;

// round-to-nearest-even f32 -> bf16 (bit-level)
__device__ __forceinline__ bf16_t f2bf(float f) {
    unsigned u = __builtin_bit_cast(unsigned, f);
    unsigned r = u + 0x7FFFu + ((u >> 16) & 1u);
    unsigned short h = (unsigned short)(r >> 16);
    return __builtin_bit_cast(bf16_t, h);
}

// float atomic max via signed-max / unsigned-min trick (init must be -inf)
__device__ __forceinline__ void atomicMaxFloat(float* addr, float val) {
    if (val >= 0.0f)
        atomicMax((int*)addr, __float_as_int(val));
    else
        atomicMin((unsigned int*)addr, (unsigned int)__float_as_int(val));
}

// ---------------------------------------------------------------------------
// GEMM: Y[M,256] = Xbf[M,256] @ W^T + b, X and W both pre-converted to bf16.
// One wave computes a 16(M) x 64(N) tile: 4 f32 accumulators, one shared A
// load per k-step, 4 x v_wmma_f32_16x16x32_bf16 per k-step (32 total).
// Block = 128 threads = 4 waves -> full 256-column coverage per m-tile.
// ---------------------------------------------------------------------------
__global__ __launch_bounds__(128) void gemm_xwT_bias(
    const bf16_t* __restrict__ Xb, const bf16_t* __restrict__ Wb,
    const float* __restrict__ bias, float* __restrict__ Y, int M)
{
    const int lane = threadIdx.x & 31;
    const int wave = threadIdx.x >> 5;      // 0..3
    const int m0   = blockIdx.x << 4;
    const int n0   = wave << 6;             // 64-col slab per wave
    if (m0 >= M) return;                    // uniform per block

    const int half = lane >> 4;             // 0: lanes 0-15, 1: lanes 16-31
    const int l15  = lane & 15;

    int mA = m0 + l15;                      // A-operand row (ISA 16-bit A layout)
    if (mA >= M) mA = M - 1;                // safe pad (M=50000 is exact)

    const bf16_t* xrow = Xb + (size_t)mA * D_MODEL;
    const bf16_t* wrow0 = Wb + (size_t)(n0 +  0 + l15) * D_MODEL + (half << 4);
    const bf16_t* wrow1 = Wb + (size_t)(n0 + 16 + l15) * D_MODEL + (half << 4);
    const bf16_t* wrow2 = Wb + (size_t)(n0 + 32 + l15) * D_MODEL + (half << 4);
    const bf16_t* wrow3 = Wb + (size_t)(n0 + 48 + l15) * D_MODEL + (half << 4);

    v8f acc0 = {0.f,0.f,0.f,0.f,0.f,0.f,0.f,0.f};
    v8f acc1 = acc0, acc2 = acc0, acc3 = acc0;

#pragma unroll
    for (int k0 = 0; k0 < D_MODEL; k0 += 32) {
        // A per lane: two contiguous 8-element (16B) runs
        const bf16_t* xa = xrow + k0 + (half << 4);
        v8bf a_lo = *(const v8bf*)(xa);
        v8bf a_hi = *(const v8bf*)(xa + 8);
        v16bf A = __builtin_shufflevector(a_lo, a_hi,
                                          0,1,2,3,4,5,6,7,8,9,10,11,12,13,14,15);

        // B per lane: fixed N, contiguous 16-element (32B) run
        v16bf B0 = *(const v16bf*)(wrow0 + k0);
        v16bf B1 = *(const v16bf*)(wrow1 + k0);
        v16bf B2 = *(const v16bf*)(wrow2 + k0);
        v16bf B3 = *(const v16bf*)(wrow3 + k0);

        acc0 = __builtin_amdgcn_wmma_f32_16x16x32_bf16(false, A, false, B0,
                                                       (short)0, acc0, false, false);
        acc1 = __builtin_amdgcn_wmma_f32_16x16x32_bf16(false, A, false, B1,
                                                       (short)0, acc1, false, false);
        acc2 = __builtin_amdgcn_wmma_f32_16x16x32_bf16(false, A, false, B2,
                                                       (short)0, acc2, false, false);
        acc3 = __builtin_amdgcn_wmma_f32_16x16x32_bf16(false, A, false, B3,
                                                       (short)0, acc3, false, false);
    }

    // C/D layout: element r of acc -> row m0 + r + 8*half, col = tile base + l15
#pragma unroll
    for (int j = 0; j < 4; ++j) {
        v8f acc = (j == 0) ? acc0 : (j == 1) ? acc1 : (j == 2) ? acc2 : acc3;
        const int col = n0 + (j << 4) + l15;
        const float bcol = bias[col];
#pragma unroll
        for (int r = 0; r < 8; ++r) {
            const int row = m0 + r + (half << 3);
            if (row < M) Y[(size_t)row * D_MODEL + col] = acc[r] + bcol;
        }
    }
}

// ---------------------------------------------------------------------------
__global__ void f32_to_bf16_kernel(const float* __restrict__ in,
                                   bf16_t* __restrict__ out, int n)
{
    int i = blockIdx.x * blockDim.x + threadIdx.x;
    if (i < n) out[i] = f2bf(in[i]);
}

__global__ void init_buffers(float* __restrict__ mbuf, float* __restrict__ den,
                             float* __restrict__ agg, int nmh, int nagg)
{
    int i = blockIdx.x * blockDim.x + threadIdx.x;
    if (i < nmh) { mbuf[i] = -__builtin_inff(); den[i] = 0.0f; }
    if (i < nagg) agg[i] = 0.0f;
}

// pass 1: per-(edge,head) scores + running segment max (atomic)
__global__ __launch_bounds__(256) void edge_scores(
    const float* __restrict__ q, const float* __restrict__ k,
    const int* __restrict__ dst, const int* __restrict__ src,
    float* __restrict__ scores, float* __restrict__ mbuf, int E)
{
    int t = blockIdx.x * blockDim.x + threadIdx.x;
    if (t >= E * N_HEADS) return;
    int e = t >> 3, h = t & 7;
    int d = dst[e], s = src[e];
    const float4* qp = (const float4*)(q + (size_t)d * D_MODEL + h * D_K);
    const float4* kp = (const float4*)(k + (size_t)s * D_MODEL + h * D_K);
    float acc = 0.0f;
#pragma unroll
    for (int i = 0; i < D_K / 4; ++i) {
        float4 a = qp[i], b = kp[i];
        acc = fmaf(a.x, b.x, acc); acc = fmaf(a.y, b.y, acc);
        acc = fmaf(a.z, b.z, acc); acc = fmaf(a.w, b.w, acc);
    }
    float sc = acc * 0.17677669529663689f;   // 1/sqrt(32)
    scores[t] = sc;
    atomicMaxFloat(mbuf + (size_t)d * N_HEADS + h, sc);
}

// pass 2: exponentiate, accumulate denominators
__global__ __launch_bounds__(256) void edge_exp(
    const int* __restrict__ dst, const float* __restrict__ mbuf,
    float* __restrict__ scores, float* __restrict__ den, int E)
{
    int t = blockIdx.x * blockDim.x + threadIdx.x;
    if (t >= E * N_HEADS) return;
    int e = t >> 3, h = t & 7;
    int d = dst[e];
    float ex = __expf(scores[t] - mbuf[(size_t)d * N_HEADS + h]);
    scores[t] = ex;                          // overwrite with numerator
    unsafeAtomicAdd(den + (size_t)d * N_HEADS + h, ex);
}

// pass 3: normalize + weighted scatter of v[src] into agg[dst]
__global__ __launch_bounds__(256) void edge_aggregate(
    const float* __restrict__ v, const int* __restrict__ dst,
    const int* __restrict__ src, const float* __restrict__ ex,
    const float* __restrict__ den, float* __restrict__ agg, int E)
{
    int e = blockIdx.x;
    if (e >= E) return;
    int t = threadIdx.x;
    int h = t >> 5;
    int d = dst[e], s = src[e];
    float p = ex[(size_t)e * N_HEADS + h] / den[(size_t)d * N_HEADS + h];
    unsafeAtomicAdd(agg + (size_t)d * D_MODEL + t, p * v[(size_t)s * D_MODEL + t]);
}

// ---------------------------------------------------------------------------
extern "C" void kernel_launch(void* const* d_in, const int* in_sizes, int n_in,
                              void* d_out, int out_size, void* d_ws, size_t ws_size,
                              hipStream_t stream)
{
    (void)n_in; (void)out_size; (void)ws_size;

    const float* query = (const float*)d_in[0];
    const float* key_  = (const float*)d_in[1];
    const float* value = (const float*)d_in[2];
    const int*   edges = (const int*)d_in[3];
    const float* Wq = (const float*)d_in[4];
    const float* bq = (const float*)d_in[5];
    const float* Wk = (const float*)d_in[6];
    const float* bk = (const float*)d_in[7];
    const float* Wv = (const float*)d_in[8];
    const float* bv = (const float*)d_in[9];
    const float* Wo = (const float*)d_in[10];
    const float* bo = (const float*)d_in[11];

    const int N = in_sizes[0] / D_MODEL;
    const int E = in_sizes[3] / 2;
    const int* dst = edges;
    const int* src = edges + E;

    // workspace carve-up (256B aligned): ~250 MB total
    char* ws = (char*)d_ws;
    size_t off = 0;
    auto salloc = [&](size_t bytes) -> void* {
        void* p = ws + off;
        off += (bytes + 255) & ~(size_t)255;
        return p;
    };
    float*  qb     = (float*)salloc((size_t)N * D_MODEL * sizeof(float));
    float*  kb     = (float*)salloc((size_t)N * D_MODEL * sizeof(float));
    float*  vb     = (float*)salloc((size_t)N * D_MODEL * sizeof(float));
    float*  agg    = (float*)salloc((size_t)N * D_MODEL * sizeof(float));
    float*  scores = (float*)salloc((size_t)E * N_HEADS * sizeof(float));
    float*  mbuf   = (float*)salloc((size_t)N * N_HEADS * sizeof(float));
    float*  den    = (float*)salloc((size_t)N * N_HEADS * sizeof(float));
    bf16_t* xbf    = (bf16_t*)salloc((size_t)N * D_MODEL * sizeof(bf16_t)); // staging
    bf16_t* Wqb    = (bf16_t*)salloc((size_t)D_MODEL * D_MODEL * sizeof(bf16_t));
    bf16_t* Wkb    = (bf16_t*)salloc((size_t)D_MODEL * D_MODEL * sizeof(bf16_t));
    bf16_t* Wvb    = (bf16_t*)salloc((size_t)D_MODEL * D_MODEL * sizeof(bf16_t));
    bf16_t* Wob    = (bf16_t*)salloc((size_t)D_MODEL * D_MODEL * sizeof(bf16_t));

    // weights -> bf16 (64K elements each)
    const int wn = D_MODEL * D_MODEL;
    f32_to_bf16_kernel<<<(wn + 255) / 256, 256, 0, stream>>>(Wq, Wqb, wn);
    f32_to_bf16_kernel<<<(wn + 255) / 256, 256, 0, stream>>>(Wk, Wkb, wn);
    f32_to_bf16_kernel<<<(wn + 255) / 256, 256, 0, stream>>>(Wv, Wvb, wn);
    f32_to_bf16_kernel<<<(wn + 255) / 256, 256, 0, stream>>>(Wo, Wob, wn);

    // init segment-max / denominators / aggregation buffers
    const int nagg = N * D_MODEL;
    init_buffers<<<(nagg + 255) / 256, 256, 0, stream>>>(mbuf, den, agg,
                                                         N * N_HEADS, nagg);

    // QKV projections: stage activations in bf16, then WMMA GEMM
    dim3 gg((N + 15) / 16);
    f32_to_bf16_kernel<<<(nagg + 255) / 256, 256, 0, stream>>>(query, xbf, nagg);
    gemm_xwT_bias<<<gg, 128, 0, stream>>>(xbf, Wqb, bq, qb, N);
    f32_to_bf16_kernel<<<(nagg + 255) / 256, 256, 0, stream>>>(key_, xbf, nagg);
    gemm_xwT_bias<<<gg, 128, 0, stream>>>(xbf, Wkb, bk, kb, N);
    f32_to_bf16_kernel<<<(nagg + 255) / 256, 256, 0, stream>>>(value, xbf, nagg);
    gemm_xwT_bias<<<gg, 128, 0, stream>>>(xbf, Wvb, bv, vb, N);

    // edge-softmax passes (L2-resident gathers, native fp32 atomics)
    const int eh = E * N_HEADS;
    edge_scores<<<(eh + 255) / 256, 256, 0, stream>>>(qb, kb, dst, src,
                                                      scores, mbuf, E);
    edge_exp<<<(eh + 255) / 256, 256, 0, stream>>>(dst, mbuf, scores, den, E);
    edge_aggregate<<<E, 256, 0, stream>>>(vb, dst, src, scores, den, agg, E);

    // output projection -> d_out
    f32_to_bf16_kernel<<<(nagg + 255) / 256, 256, 0, stream>>>(agg, xbf, nagg);
    gemm_xwT_bias<<<gg, 128, 0, stream>>>(xbf, Wob, bo, (float*)d_out, N);
}
